// CustomSupConLoss_9328668967870
// MI455X (gfx1250) — matrix operand order repaired and checked by reading, not compiled, not run on hardware
//
#include <hip/hip_runtime.h>

#define SN 8192
#define SD 512
#define TEMP 0.1f
#define INV_T 10.0f
#define NJCH 8
#define JCHUNK (SN / NJCH) /* 1024 columns per block */

typedef __attribute__((ext_vector_type(16))) __bf16 v16bf;
typedef __attribute__((ext_vector_type(8)))  __bf16 v8bf;
typedef __attribute__((ext_vector_type(8)))  float  v8f;

static __device__ __forceinline__ v8f wmma_bf16(v16bf a, v16bf b, v8f c) {
  // 8 args: (neg_a, A, neg_b, B, c_mod, C, reuse_a, reuse_b)
  return __builtin_amdgcn_wmma_f32_16x16x32_bf16(false, a, false, b, (short)0, c,
                                                 false, false);
}

// Build a 16-element bf16 fragment from two contiguous 16-byte chunks.
static __device__ __forceinline__ v16bf load_frag(const __bf16* p0, const __bf16* p1) {
  union { v16bf v; v8bf h[2]; } u;
  u.h[0] = *(const v8bf*)p0;
  u.h[1] = *(const v8bf*)p1;
  return u.v;
}

// fp32 -> (bf16 hi, bf16 lo) split, round-to-nearest-even, manual bit ops.
__global__ __launch_bounds__(256) void supcon_split(const float* __restrict__ E,
                                                    unsigned short* __restrict__ hi,
                                                    unsigned short* __restrict__ lo) {
  const size_t idx = (size_t)blockIdx.x * 256 + threadIdx.x;
  if (idx >= (size_t)SN * SD) return;
  const float x = E[idx];
  const unsigned xb = __float_as_uint(x);
  const unsigned short h = (unsigned short)((xb + 0x7fffu + ((xb >> 16) & 1u)) >> 16);
  const float hf = __uint_as_float((unsigned)h << 16);
  const float r = x - hf;
  const unsigned rb = __float_as_uint(r);
  hi[idx] = h;
  lo[idx] = (unsigned short)((rb + 0x7fffu + ((rb >> 16) & 1u)) >> 16);
}

// Each wave: 16-anchor strip, sweeps JCHUNK columns in 64-wide j-blocks.
// Online (flash-style) logsumexp + positive accumulation; partials to ws.
__global__ __launch_bounds__(256) void supcon_tiles(
    const __bf16* __restrict__ Ehi, const __bf16* __restrict__ Elo,
    const int* __restrict__ labels,
    float* __restrict__ Pm, float* __restrict__ Pl,
    float* __restrict__ Ps, float* __restrict__ Pc) {
  const int lane = threadIdx.x & 31;
  const int wave = threadIdx.x >> 5;
  const int sel  = lane >> 4;   // half-wave select
  const int ln   = lane & 15;
  const int i0   = blockIdx.x * 128 + wave * 16;
  const int jch  = blockIdx.y;
  const int jstart = jch * JCHUNK;

  // Anchor labels for this lane's 8 row slots (row = i0 + 8*sel + v).
  int labi[8];
#pragma unroll
  for (int v = 0; v < 8; ++v) labi[v] = labels[i0 + 8 * sel + v];

  float m[8], l[8], ps[8], pc[8];
#pragma unroll
  for (int v = 0; v < 8; ++v) { m[v] = -3.0e38f; l[v] = 0.f; ps[v] = 0.f; pc[v] = 0.f; }

  // A fragment: row i0+ln, chunks at kb + sel*8 and kb + 16 + sel*8.
  const __bf16* arow_hi = Ehi + (size_t)(i0 + ln) * SD;
  const __bf16* arow_lo = Elo + (size_t)(i0 + ln) * SD;
  const int aofs = sel * 8;

  for (int jb = 0; jb < JCHUNK; jb += 64) {
    const int j0 = jstart + jb;

    v8f acc[4] = {};

    for (int kb = 0; kb < SD; kb += 32) {
      v16bf ahi = load_frag(arow_hi + kb + aofs, arow_hi + kb + 16 + aofs);
      v16bf alo = load_frag(arow_lo + kb + aofs, arow_lo + kb + 16 + aofs);

      // Stage all 4 column-tiles' B fragments (hi and lo) up front so the
      // 12 WMMAs below form independent-destination bursts with no WAR
      // address math inside the hazard window.
      v16bf bhi[4], blo[4];
#pragma unroll
      for (int c = 0; c < 4; ++c) {
        // B fragment: column n = ln is row (j0 + c*16 + ln) of E;
        // lane holds K chunks at kb + sel*8 and kb + 16 + sel*8.
        const size_t brow = (size_t)(j0 + c * 16 + ln) * SD + kb + sel * 8;
        bhi[c] = load_frag(Ehi + brow, Ehi + brow + 16);
        blo[c] = load_frag(Elo + brow, Elo + brow + 16);
      }

      // Product-major order: consecutive WMMAs hit different accumulators,
      // so no back-to-back D->C dependence on the matrix pipe.
#pragma unroll
      for (int c = 0; c < 4; ++c) acc[c] = wmma_bf16(ahi, bhi[c], acc[c]); // hi*hi
#pragma unroll
      for (int c = 0; c < 4; ++c) acc[c] = wmma_bf16(ahi, blo[c], acc[c]); // hi*lo
#pragma unroll
      for (int c = 0; c < 4; ++c) acc[c] = wmma_bf16(alo, bhi[c], acc[c]); // lo*hi
    }

    int labj[4];
#pragma unroll
    for (int c = 0; c < 4; ++c) labj[c] = labels[j0 + c * 16 + ln];

#pragma unroll
    for (int v = 0; v < 8; ++v) {
      const int ig = i0 + 8 * sel + v;
      float s[4];
      float lm = -3.0e38f;
#pragma unroll
      for (int c = 0; c < 4; ++c) {
        const int jg = j0 + c * 16 + ln;
        const float sv = acc[c][v] * INV_T;
        const bool diag = (jg == ig);
        // Branchless positive accumulation: v_cndmask + add, no exec churn.
        const bool pos = (!diag) && (labj[c] == labi[v]);
        ps[v] += pos ? sv : 0.0f;
        pc[v] += pos ? 1.0f : 0.0f;
        s[c] = diag ? -3.0e38f : sv;
        lm = fmaxf(lm, s[c]);
      }
#pragma unroll
      for (int off = 1; off < 16; off <<= 1)
        lm = fmaxf(lm, __shfl_xor(lm, off, 16));
      const float nm = fmaxf(m[v], lm);
      float se = 0.f;
#pragma unroll
      for (int c = 0; c < 4; ++c) se += __expf(s[c] - nm);
#pragma unroll
      for (int off = 1; off < 16; off <<= 1)
        se += __shfl_xor(se, off, 16);
      l[v] = l[v] * __expf(m[v] - nm) + se;
      m[v] = nm;
    }
  }

  // Reduce positive partials across the 16-lane half.
#pragma unroll
  for (int v = 0; v < 8; ++v) {
#pragma unroll
    for (int off = 1; off < 16; off <<= 1) {
      ps[v] += __shfl_xor(ps[v], off, 16);
      pc[v] += __shfl_xor(pc[v], off, 16);
    }
  }

  if (ln == 0) {
#pragma unroll
    for (int v = 0; v < 8; ++v) {
      const int ig = i0 + 8 * sel + v;
      const size_t idx = (size_t)jch * SN + ig;
      Pm[idx] = m[v]; Pl[idx] = l[v]; Ps[idx] = ps[v]; Pc[idx] = pc[v];
    }
  }
}

// Merge NJCH chunk partials per anchor: exact LSE combine + positive mean.
__global__ __launch_bounds__(256) void supcon_merge(
    const float* __restrict__ Pm, const float* __restrict__ Pl,
    const float* __restrict__ Ps, const float* __restrict__ Pc,
    float* __restrict__ vout) {
  const int i = blockIdx.x * blockDim.x + threadIdx.x;
  if (i >= SN) return;
  float M = -3.0e38f;
  for (int k = 0; k < NJCH; ++k) M = fmaxf(M, Pm[(size_t)k * SN + i]);
  float L = 0.f, S = 0.f, C = 0.f;
  for (int k = 0; k < NJCH; ++k) {
    L += Pl[(size_t)k * SN + i] * __expf(Pm[(size_t)k * SN + i] - M);
    S += Ps[(size_t)k * SN + i];
    C += Pc[(size_t)k * SN + i];
  }
  vout[i] = S / C - (M + __logf(L));
}

__global__ __launch_bounds__(256) void supcon_reduce(const float* __restrict__ vin,
                                                     float* __restrict__ out) {
  __shared__ float sm[8];
  float s = 0.f;
  for (int i = threadIdx.x; i < SN; i += 256) s += vin[i];
#pragma unroll
  for (int off = 1; off < 32; off <<= 1) s += __shfl_xor(s, off, 32);
  const int lane = threadIdx.x & 31, wv = threadIdx.x >> 5;
  if (lane == 0) sm[wv] = s;
  __syncthreads();
  if (threadIdx.x == 0) {
    float t = 0.f;
    for (int w = 0; w < 8; ++w) t += sm[w];
    out[0] = -t * TEMP / (float)SN;
  }
}

extern "C" void kernel_launch(void* const* d_in, const int* in_sizes, int n_in,
                              void* d_out, int out_size, void* d_ws, size_t ws_size,
                              hipStream_t stream) {
  const float* E = (const float*)d_in[0];
  const int* labels = (const int*)d_in[1];
  float* out = (float*)d_out;

  char* ws = (char*)d_ws;
  unsigned short* hi_u = (unsigned short*)ws;                            // 8 MB
  unsigned short* lo_u = (unsigned short*)(ws + (size_t)SN * SD * 2);    // 8 MB
  float* Pm = (float*)(ws + (size_t)SN * SD * 4);                        // 256 KB each
  float* Pl = Pm + (size_t)NJCH * SN;
  float* Ps = Pl + (size_t)NJCH * SN;
  float* Pc = Ps + (size_t)NJCH * SN;
  float* vv = Pc + (size_t)NJCH * SN;                                    // 32 KB

  supcon_split<<<(SN * SD + 255) / 256, 256, 0, stream>>>(E, hi_u, lo_u);
  supcon_tiles<<<dim3(SN / 128, NJCH), 256, 0, stream>>>(
      (const __bf16*)hi_u, (const __bf16*)lo_u, labels, Pm, Pl, Ps, Pc);
  supcon_merge<<<(SN + 255) / 256, 256, 0, stream>>>(Pm, Pl, Ps, Pc, vv);
  supcon_reduce<<<1, 256, 0, stream>>>(vv, out);
}
